// NonLocalBlockND_65326452572958
// MI455X (gfx1250) — matrix operand order repaired and checked by reading, not compiled, not run on hardware
//
#include <hip/hip_runtime.h>
#include <hip/hip_bf16.h>

typedef __attribute__((ext_vector_type(16))) __bf16 v16bf;
typedef __attribute__((ext_vector_type(8)))  float  v8f;

union V16 { v16bf v; uint4 q[2]; };

#define BB 4
#define CC 256
#define NN 4096
#define CI 128

// ---------------------------------------------------------------------------
// CDNA5 async memory->LDS copy (16 bytes), tracked by ASYNCcnt.
// GVS form: LDS offset in vdst VGPR; global address = SGPR base + 32-bit VGPR offset
// ---------------------------------------------------------------------------
__device__ __forceinline__ void async_cp16(void* lds, const void* gbase, int goff) {
  unsigned ldsOff = (unsigned)(size_t)lds;   // low 32 bits of generic = LDS offset
  asm volatile("global_load_async_to_lds_b128 %0, %1, %2"
               :: "v"(ldsOff), "v"(goff), "s"(gbase) : "memory");
}
__device__ __forceinline__ void wait_async0() {
  asm volatile("s_wait_asynccnt 0x0" ::: "memory");
}

// ---------------------------------------------------------------------------
// Kernel 1: SimAM  x1 = x * sigmoid( (x-mu)^2 / (4*(sum_d2/(N-1)+eps)) + 0.5 )
// one block per (b,c) row of 4096, 256 threads
// ---------------------------------------------------------------------------
__global__ __launch_bounds__(256) void k_simam(const float* __restrict__ x,
                                               float* __restrict__ x1) {
  int bc = blockIdx.x;
  const float* row = x + (size_t)bc * NN;
  float* orow = x1 + (size_t)bc * NN;
  int t = threadIdx.x;
  float v[16]; float s = 0.f, ss = 0.f;
#pragma unroll
  for (int i = 0; i < 16; i++) { float a = row[t + 256 * i]; v[i] = a; s += a; ss += a * a; }
#pragma unroll
  for (int off = 1; off < 32; off <<= 1) { s += __shfl_xor(s, off, 32); ss += __shfl_xor(ss, off, 32); }
  __shared__ float red0[8], red1[8];
  int wid = t >> 5, lane = t & 31;
  if (lane == 0) { red0[wid] = s; red1[wid] = ss; }
  __syncthreads();
  if (t == 0) {
    float a = 0.f, b = 0.f;
#pragma unroll
    for (int i = 0; i < 8; i++) { a += red0[i]; b += red1[i]; }
    red0[0] = a; red1[0] = b;
  }
  __syncthreads();
  s = red0[0]; ss = red1[0];
  float mu = s * (1.f / (float)NN);
  float sumd2 = ss - s * mu;
  float inv = 1.f / (4.f * (sumd2 * (1.f / (float)(NN - 1)) + 1e-4f));
#pragma unroll
  for (int i = 0; i < 16; i++) {
    float d = v[i] - mu;
    float y = d * d * inv + 0.5f;
    orow[t + 256 * i] = v[i] / (1.f + __expf(-y));
  }
}

// ---------------------------------------------------------------------------
// Kernel 2: 1x1 projection  proj[ci][n] = sum_c W[ci][c]*in[b][c][n] + bias[ci]
// one wave per 16-row n-tile; WMMA: M=n(16), N=ci(16), K=c(32) x 8 chunks.
// featMajor=false -> bf16 [B][N][CI] (Q / K consumers)
// featMajor=true  -> bf16 [B][CI][N] (attention V consumer, b128-packed store)
// ---------------------------------------------------------------------------
__global__ __launch_bounds__(32) void k_proj(const float* __restrict__ in,
                                             const float* __restrict__ Wm,
                                             const float* __restrict__ bias,
                                             __bf16* __restrict__ outT,
                                             int featMajor) {
  int lane = threadIdx.x & 31;
  int L = lane & 15, hi = lane >> 4;
  int nbase = blockIdx.x * 16, b = blockIdx.y;
  const float* inb = in + (size_t)b * CC * NN;
  int n = nbase + L;

  v16bf a[8];
#pragma unroll
  for (int ch = 0; ch < 8; ch++) {
    V16 ta;
#pragma unroll
    for (int e = 0; e < 16; e++) {
      int c = ch * 32 + hi * 16 + e;
      ta.v[e] = (__bf16)inb[(size_t)c * NN + n];
    }
    a[ch] = ta.v;
  }
#pragma unroll
  for (int ct = 0; ct < 8; ct++) {
    int ci = ct * 16 + L;
    v8f acc;
#pragma unroll
    for (int i = 0; i < 8; i++) acc[i] = 0.f;
#pragma unroll
    for (int ch = 0; ch < 8; ch++) {
      const float* wr = Wm + (size_t)ci * CC + ch * 32 + hi * 16;
      V16 tb;
#pragma unroll
      for (int e = 0; e < 16; e++) tb.v[e] = (__bf16)wr[e];
      acc = __builtin_amdgcn_wmma_f32_16x16x32_bf16(false, a[ch], false, tb.v,
                                                    (short)0, acc, false, false);
    }
    float bv = bias[ci];
    if (featMajor) {
      union { uint4 q; __bf16 h[8]; } pk;
#pragma unroll
      for (int v = 0; v < 8; v++) pk.h[v] = (__bf16)(acc[v] + bv);
      // rows v are 8 consecutive n at nbase + 8*hi
      *(uint4*)&outT[((size_t)b * CI + ci) * NN + nbase + 8 * hi] = pk.q;
    } else {
#pragma unroll
      for (int v = 0; v < 8; v++) {
        int nn2 = nbase + v + 8 * hi;
        outT[((size_t)b * NN + nn2) * CI + ci] = (__bf16)(acc[v] + bv);
      }
    }
  }
}

// ---------------------------------------------------------------------------
// Kernel 3: fused flash attention  O = softmax(Q @ K^T) @ V   (per stream)
// Q,K: bf16 [B][N][CI]; V: bf16 [B][CI][N] (feature-major).
// 8 waves/block, each wave owns a 16-query tile. 32-key K/V blocks are
// double-buffered in LDS via GLOBAL_LOAD_ASYNC_TO_LDS_B128 (ASYNCcnt).
// ---------------------------------------------------------------------------
__global__ __launch_bounds__(256) void k_attn(const __bf16* __restrict__ QT,
                                              const __bf16* __restrict__ KT,
                                              const __bf16* __restrict__ Vfm,
                                              float* __restrict__ Og) {
  __shared__ __align__(16) __bf16 kS[2][32][136];   // [buf][k][c], padded rows
  __shared__ __align__(16) __bf16 vS[2][CI][40];    // [buf][ci][k], padded rows
  __shared__ __align__(16) __bf16 pS[8][16][40];    // per-wave P scratch [q][k]

  int t = threadIdx.x, wid = t >> 5, lane = t & 31;
  int L = lane & 15, hi = lane >> 4;
  int b = blockIdx.y;
  int qbase = blockIdx.x * 128 + wid * 16;

  const __bf16* Qb = QT + (size_t)b * NN * CI;
  const __bf16* Kb = KT + (size_t)b * NN * CI;
  const __bf16* Vb = Vfm + (size_t)b * CI * NN;

  // staging partition (256 threads): K tile 32x128, V tile 128x32
  int kr = t >> 3, ksg = t & 7;      // K: row kr, 16-elem segment ksg
  int vci = t >> 1, vsg = t & 1;     // V: ci row, 16-elem key segment

  auto stage = [&](int buf, int kb) {
    int gko = (int)(((size_t)(kb + kr) * CI + ksg * 16) * sizeof(__bf16));
    async_cp16(&kS[buf][kr][ksg * 16],     Kb, gko);
    async_cp16(&kS[buf][kr][ksg * 16 + 8], Kb, gko + 16);
    int gvo = (int)(((size_t)vci * NN + kb + vsg * 16) * sizeof(__bf16));
    async_cp16(&vS[buf][vci][vsg * 16],     Vb, gvo);
    async_cp16(&vS[buf][vci][vsg * 16 + 8], Vb, gvo + 16);
  };

  // Q tile in A-operand layout, 4 K-chunks of 32 channels
  v16bf qa[4];
  {
    const __bf16* qrow = Qb + (size_t)(qbase + L) * CI;
#pragma unroll
    for (int ch = 0; ch < 4; ch++) {
      V16 tq;
      tq.q[0] = *(const uint4*)(qrow + ch * 32 + hi * 8);
      tq.q[1] = *(const uint4*)(qrow + ch * 32 + 16 + hi * 8);
      qa[ch] = tq.v;
    }
  }

  float m_i[8], l_i[8];
  v8f acc[8];
#pragma unroll
  for (int v = 0; v < 8; v++) { m_i[v] = -3.0e38f; l_i[v] = 0.f; }
#pragma unroll
  for (int ct = 0; ct < 8; ct++)
#pragma unroll
    for (int v = 0; v < 8; v++) acc[ct][v] = 0.f;

  // prologue: fill buffer 0
  stage(0, 0);
  wait_async0();
  __syncthreads();

  int buf = 0;
  for (int kb = 0; kb < NN; kb += 32) {
    if (kb + 32 < NN) stage(buf ^ 1, kb + 32);  // overlap next block with compute

    // --- QK^T: two 16x16 logit tiles over 32 keys ---
    v8f lt[2];
#pragma unroll
    for (int kt = 0; kt < 2; kt++) {
      v8f s;
#pragma unroll
      for (int i = 0; i < 8; i++) s[i] = 0.f;
      const __bf16* krow = &kS[buf][kt * 16 + L][0];
#pragma unroll
      for (int ch = 0; ch < 4; ch++) {
        V16 tb;
        tb.q[0] = *(const uint4*)(krow + ch * 32 + hi * 8);
        tb.q[1] = *(const uint4*)(krow + ch * 32 + 16 + hi * 8);
        s = __builtin_amdgcn_wmma_f32_16x16x32_bf16(false, qa[ch], false, tb.v,
                                                    (short)0, s, false, false);
      }
      lt[kt] = s;
    }

    // --- online softmax update (rows live within 16-lane halves) ---
    float scv[8];
#pragma unroll
    for (int v = 0; v < 8; v++) {
      float r0 = lt[0][v], r1 = lt[1][v];
      float mx = fmaxf(r0, r1);
#pragma unroll
      for (int off = 1; off < 16; off <<= 1) mx = fmaxf(mx, __shfl_xor(mx, off, 32));
      float mn = fmaxf(m_i[v], mx);
      float sc = __expf(m_i[v] - mn);
      float p0 = __expf(r0 - mn), p1 = __expf(r1 - mn);
      float rs = p0 + p1;
#pragma unroll
      for (int off = 1; off < 16; off <<= 1) rs += __shfl_xor(rs, off, 32);
      l_i[v] = l_i[v] * sc + rs;
      m_i[v] = mn; scv[v] = sc;
      pS[wid][v + 8 * hi][L] = (__bf16)p0;        // D-layout -> LDS [q][k]
      pS[wid][v + 8 * hi][16 + L] = (__bf16)p1;
    }
#pragma unroll
    for (int ct = 0; ct < 8; ct++)
#pragma unroll
      for (int v = 0; v < 8; v++) acc[ct][v] *= scv[v];

    // reload P in A-operand layout (same-wave LDS, compiler inserts dscnt wait)
    V16 tp;
    tp.q[0] = *(const uint4*)(&pS[wid][L][hi * 16]);
    tp.q[1] = *(const uint4*)(&pS[wid][L][hi * 16 + 8]);

    // --- P @ V: 8 ci-tiles, K=32 keys ---
#pragma unroll
    for (int ct = 0; ct < 8; ct++) {
      const __bf16* vrow = &vS[buf][ct * 16 + L][hi * 16];
      V16 tv2;
      tv2.q[0] = *(const uint4*)(vrow);
      tv2.q[1] = *(const uint4*)(vrow + 8);
      acc[ct] = __builtin_amdgcn_wmma_f32_16x16x32_bf16(false, tp.v, false, tv2.v,
                                                        (short)0, acc[ct], false, false);
    }

    wait_async0();     // this wave's next-block copies landed
    __syncthreads();   // everyone done reading buf + staging buf^1
    buf ^= 1;
  }

  float* Ob = Og + (size_t)b * NN * CI;
#pragma unroll
  for (int v = 0; v < 8; v++) {
    int q = qbase + v + 8 * hi;
    float invl = 1.f / l_i[v];
#pragma unroll
    for (int ct = 0; ct < 8; ct++)
      Ob[(size_t)q * CI + ct * 16 + L] = acc[ct][v] * invl;
  }
}

// ---------------------------------------------------------------------------
// Kernel 4: z[b][co][n] = sum_ci W_out[co][ci]*(O1+O2)[b][n][ci] + b_out[co] + x
// WMMA: M=co(16), N=n(16), K=ci(32) x 4; one wave per 16-col n-tile, 16 co-tiles
// ---------------------------------------------------------------------------
__global__ __launch_bounds__(32) void k_out(const float* __restrict__ O1,
                                            const float* __restrict__ O2,
                                            const float* __restrict__ Wo,
                                            const float* __restrict__ bo,
                                            const float* __restrict__ x,
                                            float* __restrict__ z) {
  int lane = threadIdx.x & 31;
  int L = lane & 15, hi = lane >> 4;
  int nbase = blockIdx.x * 16, b = blockIdx.y;

  v16bf by[4];
  const float* o1r = O1 + ((size_t)b * NN + nbase + L) * CI;
  const float* o2r = O2 + ((size_t)b * NN + nbase + L) * CI;
#pragma unroll
  for (int ch = 0; ch < 4; ch++) {
    V16 tb;
#pragma unroll
    for (int e = 0; e < 16; e++) {
      int ci = ch * 32 + hi * 16 + e;
      tb.v[e] = (__bf16)(o1r[ci] + o2r[ci]);
    }
    by[ch] = tb.v;
  }
#pragma unroll
  for (int tt = 0; tt < 16; tt++) {
    int co = tt * 16 + L;
    const float* wr = Wo + (size_t)co * CI;
    v8f acc;
#pragma unroll
    for (int i = 0; i < 8; i++) acc[i] = 0.f;
#pragma unroll
    for (int ch = 0; ch < 4; ch++) {
      V16 ta;
#pragma unroll
      for (int e = 0; e < 16; e++) {
        int ci = ch * 32 + hi * 16 + e;
        ta.v[e] = (__bf16)wr[ci];
      }
      acc = __builtin_amdgcn_wmma_f32_16x16x32_bf16(false, ta.v, false, by[ch],
                                                    (short)0, acc, false, false);
    }
#pragma unroll
    for (int v = 0; v < 8; v++) {
      int co2 = tt * 16 + v + 8 * hi;
      int n = nbase + L;
      size_t idx = ((size_t)b * CC + co2) * NN + n;
      z[idx] = acc[v] + bo[co2] + x[idx];
    }
  }
}

// ---------------------------------------------------------------------------
extern "C" void kernel_launch(void* const* d_in, const int* in_sizes, int n_in,
                              void* d_out, int out_size, void* d_ws, size_t ws_size,
                              hipStream_t stream) {
  (void)in_sizes; (void)n_in; (void)out_size; (void)ws_size;
  const float* x       = (const float*)d_in[0];
  const float* W_theta = (const float*)d_in[1];
  const float* b_theta = (const float*)d_in[2];
  const float* W_phi   = (const float*)d_in[3];
  const float* b_phi   = (const float*)d_in[4];
  const float* W_g     = (const float*)d_in[5];
  const float* b_g     = (const float*)d_in[6];
  const float* W_D     = (const float*)d_in[7];
  const float* b_D     = (const float*)d_in[8];
  const float* W_S     = (const float*)d_in[9];
  const float* b_S     = (const float*)d_in[10];
  const float* W_out   = (const float*)d_in[11];
  const float* b_out   = (const float*)d_in[12];

  char* ws = (char*)d_ws;
  const size_t szX1   = (size_t)BB * CC * NN * sizeof(float);   // 16 MB
  const size_t szProj = (size_t)BB * NN * CI * sizeof(__bf16);  // 4 MB each
  const size_t szO    = (size_t)BB * NN * CI * sizeof(float);   // 8 MB each

  float*  x1     = (float*)ws;
  __bf16* thetaT = (__bf16*)(ws + szX1);                 // [B][N][CI]
  __bf16* phiT   = (__bf16*)(ws + szX1 + 1 * szProj);    // [B][N][CI]
  __bf16* gFM    = (__bf16*)(ws + szX1 + 2 * szProj);    // [B][CI][N]
  __bf16* DT     = (__bf16*)(ws + szX1 + 3 * szProj);    // [B][N][CI]
  __bf16* sFM    = (__bf16*)(ws + szX1 + 4 * szProj);    // [B][CI][N]
  float*  Og1    = (float*)(ws + szX1 + 5 * szProj);
  float*  Og2    = (float*)(ws + szX1 + 5 * szProj + szO);

  k_simam<<<BB * CC, 256, 0, stream>>>(x, x1);

  dim3 gp(NN / 16, BB);
  k_proj<<<gp, 32, 0, stream>>>(x,  W_theta, b_theta, thetaT, 0);
  k_proj<<<gp, 32, 0, stream>>>(x,  W_phi,   b_phi,   phiT,   0);
  k_proj<<<gp, 32, 0, stream>>>(x,  W_g,     b_g,     gFM,    1);
  k_proj<<<gp, 32, 0, stream>>>(x1, W_D,     b_D,     DT,     0);
  k_proj<<<gp, 32, 0, stream>>>(x1, W_S,     b_S,     sFM,    1);

  dim3 ga(NN / 128, BB);
  k_attn<<<ga, 256, 0, stream>>>(thetaT, phiT, gFM, Og1);
  k_attn<<<ga, 256, 0, stream>>>(DT,     phiT, sFM, Og2);

  dim3 go(NN / 16, BB);
  k_out<<<go, 32, 0, stream>>>(Og1, Og2, W_out, b_out, x, (float*)d_out);
}
